// Hybrid_QCNN_TQ_65266323030226
// MI455X (gfx1250) — matrix operand order repaired and checked by reading, not compiled, not run on hardware
//
#include <hip/hip_runtime.h>
#include <hip/hip_bf16.h>
#include <math.h>

typedef __attribute__((ext_vector_type(16))) _Float16 v16h;
typedef __attribute__((ext_vector_type(8)))  float    v8f;
typedef __attribute__((ext_vector_type(4)))  float    v4f;

union F16Frag { v16h h; v4f f[2]; };

#if defined(__AMDGCN__) && __has_builtin(__builtin_amdgcn_tensor_load_to_lds)
#define QCNN_HAVE_TDM 1
#else
#define QCNN_HAVE_TDM 0
#endif

// ---------------------------------------------------------------------------
// 16x16 WMMA tile GEMM helper.
// A fragment layout (16-bit A 16x32, ISA 7.12.2):
//   lane L: M = L&15; half h: K = 16*(h>>3) + 8*(L>=16) + (h&7)
//   -> two contiguous 8-half (16B) groups per chunk.
// B fragment layout (16-bit B 32x16): lane L: N = L&15; half h:
//   K = 16*(h>>3) + 8*(L>=16) + (h&7)  -> same as A.
// B matrices are stored row-major [n][k] (PyTorch (out,in) order), so no
// transpose is needed for the FC weights.
// ---------------------------------------------------------------------------
__device__ __forceinline__ v8f tile_gemm16(const _Float16* aRow,
                                           const _Float16* bRow,
                                           int kChunks, bool hi) {
  const int aoff = hi ? 8 : 0;
  const int boff = hi ? 8 : 0;
  v8f acc = {0.f, 0.f, 0.f, 0.f, 0.f, 0.f, 0.f, 0.f};
  for (int kc = 0; kc < kChunks; ++kc) {
    const int kb = kc * 32;
    F16Frag a, b;
    a.f[0] = *(const v4f*)(aRow + kb + aoff);
    a.f[1] = *(const v4f*)(aRow + kb + aoff + 16);
    b.f[0] = *(const v4f*)(bRow + kb + boff);
    b.f[1] = *(const v4f*)(bRow + kb + boff + 16);
    acc = __builtin_amdgcn_wmma_f32_16x16x32_f16(false, a.h, false, b.h,
                                                 (short)0, acc, false, false);
  }
  return acc;
}

// ---------------------------------------------------------------------------
// Kernel W: one-time weight conversion/repack to f16 (40960 elements).
//   wc2 [16][576]  k = (ky*3+kx)*64 + ci      (offset 0)
//   w1p [64][224]  zero-padded K 196->224     (offset 9216)
//   w2p [128][64]                              (offset 23552)
//   w3p [64][128]                              (offset 31744)
//   w4p [16][64]   rows 10..15 zero            (offset 39936)
// ---------------------------------------------------------------------------
__global__ void prep_weights(const float* __restrict__ c2w,
                             const float* __restrict__ f1w,
                             const float* __restrict__ f2w,
                             const float* __restrict__ f3w,
                             const float* __restrict__ f4w,
                             _Float16* __restrict__ wbuf) {
  const int i = blockIdx.x * 256 + threadIdx.x;
  if (i < 9216) {
    const int o = i / 576, k = i - o * 576;
    const int tap = k >> 6, ci = k & 63;
    wbuf[i] = (_Float16)c2w[o * 576 + ci * 9 + tap];
  } else if (i < 9216 + 14336) {
    const int j = i - 9216;
    const int o = j / 224, k = j - o * 224;
    wbuf[9216 + j] = (k < 196) ? (_Float16)f1w[o * 196 + k] : (_Float16)0.f;
  } else if (i < 23552 + 8192) {
    const int j = i - 23552;
    wbuf[23552 + j] = (_Float16)f2w[j];
  } else if (i < 31744 + 8192) {
    const int j = i - 31744;
    wbuf[31744 + j] = (_Float16)f3w[j];
  } else if (i < 39936 + 1024) {
    const int j = i - 39936;
    const int o = j >> 6, k = j & 63;
    wbuf[39936 + j] = (o < 10) ? (_Float16)f4w[o * 64 + k] : (_Float16)0.f;
  }
}

// ---------------------------------------------------------------------------
// Kernel A: conv1 (1->64, 3x3 SAME on 28x28) + bias + ReLU + 2x2 maxpool.
// One block per image. Output act1: (B,14,14,64) NHWC f16.
// ---------------------------------------------------------------------------
__global__ void __launch_bounds__(256) conv1_kernel(const float* __restrict__ x,
                                                    const float* __restrict__ w1,
                                                    const float* __restrict__ b1,
                                                    _Float16* __restrict__ act1) {
  __shared__ float imgs[784];
  __shared__ float w1s[576];
  __shared__ float b1s[64];
  const int tid = threadIdx.x;
  const int bimg = blockIdx.x;
  const float* img = x + (size_t)bimg * 784;
  for (int i = tid; i < 784; i += 256) imgs[i] = img[i];
  for (int i = tid; i < 576; i += 256) w1s[i] = w1[i];
  if (tid < 64) b1s[tid] = b1[tid];
  __syncthreads();
  for (int idx = tid; idx < 12544; idx += 256) {
    const int oc = idx & 63;
    const int pos = idx >> 6;
    const int ox = pos % 14, oy = pos / 14;
    float mx = -1e30f;
#pragma unroll
    for (int dy = 0; dy < 2; ++dy) {
#pragma unroll
      for (int dx = 0; dx < 2; ++dx) {
        const int py = 2 * oy + dy, px = 2 * ox + dx;
        float acc = b1s[oc];
#pragma unroll
        for (int ky = 0; ky < 3; ++ky) {
          const int yy = py + ky - 1;
          if (yy < 0 || yy >= 28) continue;
#pragma unroll
          for (int kx = 0; kx < 3; ++kx) {
            const int xx = px + kx - 1;
            if (xx < 0 || xx >= 28) continue;
            acc = fmaf(imgs[yy * 28 + xx], w1s[oc * 9 + ky * 3 + kx], acc);
          }
        }
        mx = fmaxf(mx, acc);
      }
    }
    mx = fmaxf(mx, 0.f);
    act1[(size_t)bimg * 12544 + idx] = (_Float16)mx;
  }
}

// ---------------------------------------------------------------------------
// Kernel B: conv2 as implicit GEMM via WMMA.
// M = B*196 output positions, N = 16 channels, K = 9*64 = 576 (18 chunks).
// One wave per 16x16 tile. A tile (zero-padded im2col) staged in LDS.
// The 18KB f16 weight matrix is DMA'd into LDS by the Tensor Data Mover
// (overlapping the im2col gather); B fragments then come from LDS.
// Fused bias + ReLU; output act2: (B,196,16) f16.
// ---------------------------------------------------------------------------
__global__ void __launch_bounds__(32) conv2_wmma_kernel(
    const _Float16* __restrict__ act1, const _Float16* __restrict__ wc2,
    const float* __restrict__ b2, _Float16* __restrict__ act2, int Mtotal) {
  // A tile first, weights LAST: any TDM over-iteration past the weight
  // region falls outside the LDS allocation and is dropped (ISA: OOB async
  // LDS writes are discarded) instead of clobbering the A tile.
  __shared__ _Float16 smem[16 * 584 + 16 * 576];
  _Float16* At = smem;                 // row stride 584 (pad vs bank conflicts)
  _Float16* Wlds = smem + 16 * 584;    // 16 x 576 f16 = 18432 B
  const int lane = threadIdx.x;
  const int tileM = blockIdx.x * 16;

#if QCNN_HAVE_TDM
  {
    // Tensor DMA descriptor (ISA 8.3/8.4): flat copy of 18432 B as one row
    // of 2304 8-byte elements. data_size=3 (8B), type=2 ("image").
    typedef __attribute__((ext_vector_type(4))) unsigned int tdm_v4u;
    typedef __attribute__((ext_vector_type(8))) int tdm_v8i;
    typedef __attribute__((ext_vector_type(4))) int tdm_v4i;
    const unsigned long long ga = (unsigned long long)(const void*)wc2;
    const unsigned ldsAddr = (unsigned)(unsigned long long)(void*)Wlds;
    const unsigned units = (16u * 576u * 2u) / 8u;  // 2304
    tdm_v4u g0;
    g0[0] = 1u;                                        // count=1 (valid user D#)
    g0[1] = ldsAddr;                                   // lds_addr
    g0[2] = (unsigned)(ga & 0xFFFFFFFFu);              // global_addr[31:0]
    g0[3] = (unsigned)((ga >> 32) & 0x01FFFFFFu) |     // global_addr[56:32]
            0x80000000u;                               // type=2
    tdm_v8i g1;
    g1[0] = 0x00030000;                                // data_size=3, mask=0
    g1[1] = (int)((units & 0xFFFFu) << 16);            // tensor_dim0[15:0]
    g1[2] = (int)((units >> 16) & 0xFFFFu) | (1 << 16);// dim0[31:16] | dim1=1
    g1[3] = (int)((units & 0xFFFFu) << 16);            // tile_dim0
    g1[4] = 1;                                         // tile_dim1=1, tile_dim2=0
    g1[5] = (int)units;                                // tensor_dim0_stride[31:0]
    g1[6] = 0;                                         // stride0[47:32], stride1 lo
    g1[7] = 0;
    tdm_v4i g2 = {0, 0, 0, 0};
    tdm_v4i g3 = {0, 0, 0, 0};
    tdm_v8i g4 = {0, 0, 0, 0, 0, 0, 0, 0};             // 6-arg form: extra group
    __builtin_amdgcn_tensor_load_to_lds(g0, g1, g2, g3, g4, 0);
  }
#else
  for (int i = lane; i < (16 * 576) / 8; i += 32)
    ((v4f*)Wlds)[i] = ((const v4f*)wc2)[i];
#endif

  const v4f z4 = {0.f, 0.f, 0.f, 0.f};
  // Gather: 16 rows x 9 taps, 64 f16 (128B) per (row,tap). Overlaps the TDM.
  for (int idx = lane; idx < 144; idx += 32) {
    const int ml = idx / 9, tap = idx - ml * 9;
    const int m = tileM + ml;
    _Float16* dst = At + ml * 584 + tap * 64;
    const _Float16* src = nullptr;
    if (m < Mtotal) {
      const int bimg = m / 196, pos = m - bimg * 196;
      const int y = pos / 14, xq = pos - y * 14;
      const int ky = tap / 3, kx = tap - ky * 3;
      const int yy = y + ky - 1, xx = xq + kx - 1;
      if (yy >= 0 && yy < 14 && xx >= 0 && xx < 14)
        src = act1 + (((size_t)bimg * 14 + yy) * 14 + xx) * 64;
    }
    if (src) {
#pragma unroll
      for (int q = 0; q < 8; ++q) ((v4f*)dst)[q] = ((const v4f*)src)[q];
    } else {
#pragma unroll
      for (int q = 0; q < 8; ++q) ((v4f*)dst)[q] = z4;
    }
  }
  __syncthreads();
#if QCNN_HAVE_TDM
  __builtin_amdgcn_s_wait_tensorcnt(0);  // weight tile resident in LDS
#endif
  const bool hi = lane >= 16;
  const int nn = lane & 15;
  const v8f acc = tile_gemm16(At + nn * 584, Wlds + nn * 576, 18, hi);
  // Epilogue: C/D layout -> (m = r + 8*(lane>=16), n = lane&15)
  const float bv = b2[nn];
  const int mhi = hi ? 8 : 0;
#pragma unroll
  for (int r = 0; r < 8; ++r) {
    const int m = tileM + r + mhi;
    if (m < Mtotal) {
      const float v = fmaxf(acc[r] + bv, 0.f);
      act2[(size_t)m * 16 + nn] = (_Float16)v;
    }
  }
}

// ---------------------------------------------------------------------------
// Kernel C: 2x2 maxpool -> conv3 (16->4, 3x3 SAME on 7x7) -> leaky ReLU
//           -> sigmoid * pi. One block per image. angles: (B*49, 4) f32.
// ---------------------------------------------------------------------------
__global__ void __launch_bounds__(64) pool_conv3_kernel(
    const _Float16* __restrict__ act2, const float* __restrict__ w3,
    const float* __restrict__ b3, float* __restrict__ angles) {
  __shared__ _Float16 pl[49 * 16];
  __shared__ float w3s[576];
  __shared__ float b3s[4];
  const int tid = threadIdx.x;
  const int bimg = blockIdx.x;
  for (int i = tid; i < 576; i += 64) w3s[i] = w3[i];
  if (tid < 4) b3s[tid] = b3[tid];
  const size_t base = (size_t)bimg * 3136;
  for (int idx = tid; idx < 784; idx += 64) {
    const int c = idx & 15, pos = idx >> 4;
    const int px = pos % 7, py = pos / 7;
    float mx = 0.f;  // inputs already ReLU'd (>= 0)
#pragma unroll
    for (int dy = 0; dy < 2; ++dy)
#pragma unroll
      for (int dx = 0; dx < 2; ++dx)
        mx = fmaxf(mx, (float)act2[base + ((2 * py + dy) * 14 + (2 * px + dx)) * 16 + c]);
    pl[pos * 16 + c] = (_Float16)mx;
  }
  __syncthreads();
  const float PI = 3.14159265358979f;
  for (int idx = tid; idx < 196; idx += 64) {
    const int oc = idx & 3, pos = idx >> 2;
    const int px = pos % 7, py = pos / 7;
    float acc = b3s[oc];
#pragma unroll
    for (int ky = 0; ky < 3; ++ky) {
      const int yy = py + ky - 1;
      if (yy < 0 || yy >= 7) continue;
#pragma unroll
      for (int kx = 0; kx < 3; ++kx) {
        const int xx = px + kx - 1;
        if (xx < 0 || xx >= 7) continue;
#pragma unroll
        for (int ci = 0; ci < 16; ++ci)
          acc = fmaf((float)pl[(yy * 7 + xx) * 16 + ci],
                     w3s[oc * 144 + ci * 9 + ky * 3 + kx], acc);
      }
    }
    acc = (acc > 0.f) ? acc : 0.01f * acc;                 // leaky ReLU
    const float ang = PI / (1.f + __expf(-acc));           // sigmoid * pi
    angles[((size_t)bimg * 49 + pos) * 4 + oc] = ang;
  }
}

// ---------------------------------------------------------------------------
// Kernel D: 4-qubit circuit, one patch per lane; full state (16 complex
// amplitudes) in registers. Wire indices are template constants so the
// unrolled gate ops constant-fold (no dynamic indexing / scratch).
// Wire w maps to amplitude bit (3 - w).
// ---------------------------------------------------------------------------
struct QS { float re[16]; float im[16]; };

template <int W> __device__ __forceinline__ void ryg(QS& s, float c, float sn) {
  constexpr int m = 1 << (3 - W);
#pragma unroll
  for (int i = 0; i < 16; ++i) {
    if (!(i & m)) {
      const int j = i | m;
      const float r0 = s.re[i], i0 = s.im[i], r1 = s.re[j], i1 = s.im[j];
      s.re[i] = c * r0 - sn * r1; s.im[i] = c * i0 - sn * i1;
      s.re[j] = sn * r0 + c * r1; s.im[j] = sn * i0 + c * i1;
    }
  }
}
template <int W> __device__ __forceinline__ void rzg(QS& s, float c, float sn) {
  constexpr int m = 1 << (3 - W);
#pragma unroll
  for (int i = 0; i < 16; ++i) {
    const float r = s.re[i], q = s.im[i];
    if (i & m) { s.re[i] = r * c - q * sn; s.im[i] = q * c + r * sn; }  // e^{+i t/2}
    else       { s.re[i] = r * c + q * sn; s.im[i] = q * c - r * sn; }  // e^{-i t/2}
  }
}
template <int C, int T> __device__ __forceinline__ void cnotg(QS& s) {
  constexpr int cm = 1 << (3 - C), tm = 1 << (3 - T);
#pragma unroll
  for (int i = 0; i < 16; ++i) {
    if ((i & cm) && !(i & tm)) {
      const int j = i | tm;
      const float r = s.re[i], q = s.im[i];
      s.re[i] = s.re[j]; s.im[i] = s.im[j];
      s.re[j] = r;       s.im[j] = q;
    }
  }
}
template <int A, int B> __device__ __forceinline__ void ansatz(QS& s, const float* p) {
  float sn, c;
  __sincosf(p[0] * 0.5f, &sn, &c); ryg<A>(s, c, sn);
  __sincosf(p[1] * 0.5f, &sn, &c); rzg<A>(s, c, sn);
  __sincosf(p[2] * 0.5f, &sn, &c); ryg<B>(s, c, sn);
  __sincosf(p[3] * 0.5f, &sn, &c); rzg<B>(s, c, sn);
  cnotg<A, B>(s);
  __sincosf(p[4] * 0.5f, &sn, &c); ryg<A>(s, c, sn);
  __sincosf(p[5] * 0.5f, &sn, &c); rzg<A>(s, c, sn);
  __sincosf(p[6] * 0.5f, &sn, &c); ryg<B>(s, c, sn);
}

__global__ void __launch_bounds__(256) qcircuit_kernel(
    const float* __restrict__ angles, const float* __restrict__ qp,
    float* __restrict__ qout, int P) {
  const int pid = blockIdx.x * 256 + threadIdx.x;
  if (pid >= P) return;
  float qpl[14];
#pragma unroll
  for (int i = 0; i < 14; ++i) qpl[i] = qp[i];
  QS s;
#pragma unroll
  for (int i = 0; i < 16; ++i) { s.re[i] = 0.f; s.im[i] = 0.f; }
  s.re[0] = 1.f;
  float sn, c;
  __sincosf(angles[pid * 4 + 0] * 0.5f, &sn, &c); ryg<0>(s, c, sn);
  __sincosf(angles[pid * 4 + 1] * 0.5f, &sn, &c); ryg<1>(s, c, sn);
  __sincosf(angles[pid * 4 + 2] * 0.5f, &sn, &c); ryg<2>(s, c, sn);
  __sincosf(angles[pid * 4 + 3] * 0.5f, &sn, &c); ryg<3>(s, c, sn);
#pragma unroll
  for (int layer = 0; layer < 2; ++layer) {
    const float* p = qpl + layer * 7;
    ansatz<0, 1>(s, p); ansatz<1, 2>(s, p);
    ansatz<2, 3>(s, p); ansatz<3, 0>(s, p);
  }
  float ez0 = 0.f, ez1 = 0.f, ez2 = 0.f, ez3 = 0.f;
#pragma unroll
  for (int i = 0; i < 16; ++i) {
    const float pr = s.re[i] * s.re[i] + s.im[i] * s.im[i];
    ez0 += (i & 8) ? -pr : pr;
    ez1 += (i & 4) ? -pr : pr;
    ez2 += (i & 2) ? -pr : pr;
    ez3 += (i & 1) ? -pr : pr;
  }
  const int bimg = pid / 49, pp = pid - bimg * 49;
  float* o = qout + (size_t)bimg * 196 + pp * 4;
  o[0] = ez0; o[1] = ez1; o[2] = ez2; o[3] = ez3;
}

// ---------------------------------------------------------------------------
// Kernel E: fused 4-layer MLP via WMMA. 4 waves/block; each wave owns a
// 16-row batch tile and chains layers through per-wave LDS ping-pong buffers.
//   L1: K=224(pad of 196), N=64, ReLU -> L2: K=64, N=128 ->
//   L3: K=128, N=64, ReLU -> L4: K=64, N=16 (10 valid) -> global f32.
// Requires Bn % 64 == 0 (true for B=2048).
// ---------------------------------------------------------------------------
__global__ void __launch_bounds__(128) mlp_wmma_kernel(
    const float* __restrict__ qout,
    const _Float16* __restrict__ w1p, const _Float16* __restrict__ w2p,
    const _Float16* __restrict__ w3p, const _Float16* __restrict__ w4p,
    const float* __restrict__ b1, const float* __restrict__ b2,
    const float* __restrict__ b3, const float* __restrict__ b4,
    float* __restrict__ out) {
  __shared__ _Float16 ebuf[4][16 * 224 + 16 * 128];
  const int lane = threadIdx.x & 31;
  const int wv = threadIdx.x >> 5;
  const bool hi = lane >= 16;
  const int mhi = hi ? 8 : 0;
  const int nn = lane & 15;
  const int mbase = (blockIdx.x * 4 + wv) * 16;
  _Float16* bufA = ebuf[wv];
  _Float16* bufB = bufA + 16 * 224;
  // Stage 0: 16 x 196 f32 -> f16 LDS, zero-pad K to 224.
  for (int idx = lane; idx < 16 * 224; idx += 32) {
    const int m = idx / 224, k = idx - m * 224;
    const float v = (k < 196) ? qout[(size_t)(mbase + m) * 196 + k] : 0.f;
    bufA[idx] = (_Float16)v;
  }
  __syncthreads();
  // L1: 196->64, ReLU.  bufA(stride 224) -> bufB(stride 64)
#pragma unroll
  for (int nt = 0; nt < 4; ++nt) {
    const v8f acc = tile_gemm16(bufA + nn * 224, w1p + (nt * 16 + nn) * 224, 7, hi);
    const float bv = b1[nt * 16 + nn];
#pragma unroll
    for (int r = 0; r < 8; ++r)
      bufB[(r + mhi) * 64 + nt * 16 + nn] = (_Float16)fmaxf(acc[r] + bv, 0.f);
  }
  __syncthreads();
  // L2: 64->128, linear.  bufB(64) -> bufA(stride 128)
#pragma unroll
  for (int nt = 0; nt < 8; ++nt) {
    const v8f acc = tile_gemm16(bufB + nn * 64, w2p + (nt * 16 + nn) * 64, 2, hi);
    const float bv = b2[nt * 16 + nn];
#pragma unroll
    for (int r = 0; r < 8; ++r)
      bufA[(r + mhi) * 128 + nt * 16 + nn] = (_Float16)(acc[r] + bv);
  }
  __syncthreads();
  // L3: 128->64, ReLU.  bufA(128) -> bufB(64)
#pragma unroll
  for (int nt = 0; nt < 4; ++nt) {
    const v8f acc = tile_gemm16(bufA + nn * 128, w3p + (nt * 16 + nn) * 128, 4, hi);
    const float bv = b3[nt * 16 + nn];
#pragma unroll
    for (int r = 0; r < 8; ++r)
      bufB[(r + mhi) * 64 + nt * 16 + nn] = (_Float16)fmaxf(acc[r] + bv, 0.f);
  }
  __syncthreads();
  // L4: 64->10 (N padded to 16; weight rows 10..15 are zero).
  {
    const v8f acc = tile_gemm16(bufB + nn * 64, w4p + nn * 64, 2, hi);
    if (nn < 10) {
      const float bv = b4[nn];
#pragma unroll
      for (int r = 0; r < 8; ++r)
        out[(size_t)(mbase + r + mhi) * 10 + nn] = acc[r] + bv;
    }
  }
}

// ---------------------------------------------------------------------------
extern "C" void kernel_launch(void* const* d_in, const int* in_sizes, int n_in,
                              void* d_out, int out_size, void* d_ws, size_t ws_size,
                              hipStream_t stream) {
  const float* x   = (const float*)d_in[0];
  const float* c1w = (const float*)d_in[1];
  const float* c1b = (const float*)d_in[2];
  const float* c2w = (const float*)d_in[3];
  const float* c2b = (const float*)d_in[4];
  const float* c3w = (const float*)d_in[5];
  const float* c3b = (const float*)d_in[6];
  const float* qp  = (const float*)d_in[7];
  const float* f1w = (const float*)d_in[8];
  const float* f1b = (const float*)d_in[9];
  const float* f2w = (const float*)d_in[10];
  const float* f2b = (const float*)d_in[11];
  const float* f3w = (const float*)d_in[12];
  const float* f3b = (const float*)d_in[13];
  const float* f4w = (const float*)d_in[14];
  const float* f4b = (const float*)d_in[15];
  float* out = (float*)d_out;

  const int Bn = in_sizes[0] / 784;  // 2048

  // Workspace layout (all intermediates fit in the 192MB L2):
  char* ws = (char*)d_ws;
  _Float16* wbuf = (_Float16*)ws;                              // 40960 f16 = 80KB
  _Float16* act1 = (_Float16*)(ws + 81920);                    // (B,14,14,64) f16
  const size_t o2 = 81920 + (size_t)Bn * 12544 * 2;
  _Float16* act2 = (_Float16*)(ws + o2);                       // (B,196,16) f16
  const size_t o3 = o2 + (size_t)Bn * 3136 * 2;
  float* angles = (float*)(ws + o3);                           // (B*49,4) f32
  const size_t o4 = o3 + (size_t)Bn * 196 * 4;
  float* qout = (float*)(ws + o4);                             // (B,196) f32

  prep_weights<<<160, 256, 0, stream>>>(c2w, f1w, f2w, f3w, f4w, wbuf);
  conv1_kernel<<<Bn, 256, 0, stream>>>(x, c1w, c1b, act1);
  const int Mtotal = Bn * 196;
  conv2_wmma_kernel<<<(Mtotal + 15) / 16, 32, 0, stream>>>(act1, wbuf, c2b, act2, Mtotal);
  pool_conv3_kernel<<<Bn, 64, 0, stream>>>(act2, c3w, c3b, angles);
  const int P = Bn * 49;
  qcircuit_kernel<<<(P + 255) / 256, 256, 0, stream>>>(angles, qp, qout, P);
  mlp_wmma_kernel<<<Bn / 64, 128, 0, stream>>>(qout,
      wbuf + 9216, wbuf + 23552, wbuf + 31744, wbuf + 39936,
      f1b, f2b, f3b, f4b, out);
}
